// DotProdAttention_70050916598005
// MI455X (gfx1250) — matrix operand order repaired and checked by reading, not compiled, not run on hardware
//
#include <hip/hip_runtime.h>

typedef __attribute__((ext_vector_type(16))) __bf16 v16bf;
typedef __attribute__((ext_vector_type(8)))  __bf16 v8bf;
typedef __attribute__((ext_vector_type(8)))  float  v8f;
typedef __attribute__((ext_vector_type(4)))  float  v4f;

#define S_LEN 2048
#define D_DIM 64
#define BK 64                 // keys per iteration (2 WMMA K-steps of 32)
#define NKB (S_LEN / BK)      // 32 key blocks
#define WAVES 8
#define KVS  80               // kv_lds  row stride (elems) -> 160B rows (32B aligned)
#define KVTS 80               // kvT_lds row stride (elems) -> 160B rows (32B aligned)
#define PS   80               // p_lds   row stride (elems) -> 160B rows (32B aligned)

// ---- DPP16 butterfly helpers: reduce over each 16-lane half of the wave ----
// 0xB1 = quad_perm(1,0,3,2) -> xor1 ; 0x4E = quad_perm(2,3,0,1) -> xor2
// 0x141 = row_half_mirror -> xor4 (quads uniform) ; 0x140 = row_mirror -> xor8
template <int CTRL>
__device__ __forceinline__ float dpp_movf(float x) {
    int i = __builtin_amdgcn_update_dpp(0, __builtin_bit_cast(int, x),
                                        CTRL, 0xf, 0xf, true);
    return __builtin_bit_cast(float, i);
}
__device__ __forceinline__ float rowmax16(float x) {
    x = fmaxf(x, dpp_movf<0xB1>(x));
    x = fmaxf(x, dpp_movf<0x4E>(x));
    x = fmaxf(x, dpp_movf<0x141>(x));
    x = fmaxf(x, dpp_movf<0x140>(x));
    return x;
}
__device__ __forceinline__ float rowsum16(float x) {
    x += dpp_movf<0xB1>(x);
    x += dpp_movf<0x4E>(x);
    x += dpp_movf<0x141>(x);
    x += dpp_movf<0x140>(x);
    return x;
}

__global__ __launch_bounds__(256)
void fattn_fwd_kernel(const float* __restrict__ q,
                      const float* __restrict__ kv,
                      const float* __restrict__ mask,
                      float* __restrict__ out)
{
    // double-buffered KV tile (row-major + transposed layouts)
    __shared__ alignas(32) __bf16 kv_lds [2][BK][KVS];       // [buf][key][d]
    __shared__ alignas(32) __bf16 kvT_lds[2][D_DIM][KVTS];   // [buf][d][key]
    __shared__ alignas(32) __bf16 p_lds  [WAVES][16][PS];    // per-wave P staging

    const int tid  = threadIdx.x;
    const int wave = tid >> 5;
    const int lane = tid & 31;
    const int lan  = lane & 15;
    const int hi   = lane >> 4;            // which 16-lane half
    const int bh   = blockIdx.x >> 4;      // 16 q-tiles of 128 rows each
    const int qt   = blockIdx.x & 15;
    const int q0   = qt * 128 + wave * 16; // first query row owned by this wave

    const size_t head_off = (size_t)bh * S_LEN * D_DIM;
    const float* qg  = q   + head_off;
    const float* kvg = kv  + head_off;
    float*       og  = out + head_off;

    // cooperative-load geometry: thread owns 16 consecutive floats of the tile
    const int ld_r = tid >> 2;             // tile row (0..63)
    const int ld_c = (tid & 3) * 16;       // tile col base (0/16/32/48)

    // ---- Q fragments in WMMA A-layout (fp32 -> bf16), 2 chunks over D ----
    // A 16x32 bf16: lane holds row M=lane%16; element e -> K = 16*hi + e
    v16bf qfrag[2];
    {
        const size_t rbase = (size_t)(q0 + lan) * D_DIM;
        const int    kb8   = hi * 16;
#pragma unroll
        for (int ks = 0; ks < 2; ++ks)
#pragma unroll
            for (int e = 0; e < 16; ++e) {
                int d = ks * 32 + kb8 + e;
                qfrag[ks][e] = (__bf16)qg[rbase + d];
            }
    }

    // ---- running softmax state: lane holds rows 8*hi .. 8*hi+7 ----
    float m[8], l[8];
    const v8f vzero = {0.f,0.f,0.f,0.f,0.f,0.f,0.f,0.f};
    v8f acc[4];
#pragma unroll
    for (int v = 0; v < 8; ++v) { m[v] = -1.0e30f; l[v] = 0.0f; }
#pragma unroll
    for (int t = 0; t < 4; ++t) acc[t] = vzero;

    const float c1 = 0.125f * 1.44269504088896340736f;   // (1/sqrt(D)) * log2(e)
    const float c2 = 1.44269504088896340736f;            // log2(e)

    // ---- prologue: stage tile 0 into buffer 0 ----
    {
        const v4f* src = (const v4f*)kvg;
        v16bf hv;
#pragma unroll
        for (int j = 0; j < 4; ++j) {
            v4f x = src[tid * 4 + j];
#pragma unroll
            for (int e = 0; e < 4; ++e) hv[j * 4 + e] = (__bf16)x[e];
        }
        *(v16bf*)&kv_lds[0][ld_r][ld_c] = hv;
#pragma unroll
        for (int e = 0; e < 16; ++e) kvT_lds[0][ld_c + e][ld_r] = hv[e];
    }
    __syncthreads();

    for (int kb = 0; kb < NKB; ++kb) {
        const int cur = kb & 1;
        const bool have_next = (kb + 1 < NKB);

        // issue next tile's global loads early: latency hides under the WMMAs
        v4f nxt[4];
        if (have_next) {
            const v4f* src = (const v4f*)(kvg + (size_t)(kb + 1) * BK * D_DIM);
#pragma unroll
            for (int j = 0; j < 4; ++j) nxt[j] = src[tid * 4 + j];
            if (kb + 2 < NKB)   // warm L2 two tiles ahead
                __builtin_prefetch(kvg + (size_t)(kb + 2) * BK * D_DIM + tid * 16, 0, 0);
        }

        // ---- scores: S = Q @ KV^T, four 16x16 n-tiles, K=64 over 2 steps ----
        v8f sc[4];
#pragma unroll
        for (int nt = 0; nt < 4; ++nt) {
            v8f c = vzero;
#pragma unroll
            for (int ks = 0; ks < 2; ++ks) {
                // B 32x16 bf16: lane holds col N=lane%16, element e -> K = 16*hi + e
                v16bf b = *(const v16bf*)&kv_lds[cur][nt * 16 + lan][ks * 32 + 16 * hi];
                c = __builtin_amdgcn_wmma_f32_16x16x32_bf16(
                        false, qfrag[ks], false, b, (short)0, c, false, false);
            }
            sc[nt] = c;
        }

        // ---- online softmax over these 64 keys (exp2 domain, DPP reductions) ----
        {
            float rmax[8];
#pragma unroll
            for (int v = 0; v < 8; ++v) {
                size_t moff = (size_t)(q0 + v + 8 * hi) * S_LEN + (size_t)kb * BK + lan;
#pragma unroll
                for (int nt = 0; nt < 4; ++nt)
                    sc[nt][v] = fmaf(sc[nt][v], c1, mask[moff + nt * 16] * c2);
                rmax[v] = rowmax16(fmaxf(fmaxf(sc[0][v], sc[1][v]),
                                         fmaxf(sc[2][v], sc[3][v])));
            }
#pragma unroll
            for (int v = 0; v < 8; ++v) {
                float mn    = fmaxf(m[v], rmax[v]);
                float alpha = __builtin_amdgcn_exp2f(m[v] - mn);
#pragma unroll
                for (int nt = 0; nt < 4; ++nt)
                    sc[nt][v] = __builtin_amdgcn_exp2f(sc[nt][v] - mn);  // P
                l[v] = l[v] * alpha +
                       rowsum16((sc[0][v] + sc[1][v]) + (sc[2][v] + sc[3][v]));
                m[v] = mn;
#pragma unroll
                for (int t = 0; t < 4; ++t) acc[t][v] *= alpha;
            }
        }

        // ---- stage P (bf16) row-major in wave-private LDS, reload A-layout ----
#pragma unroll
        for (int v = 0; v < 8; ++v)
#pragma unroll
            for (int nt = 0; nt < 4; ++nt)
                p_lds[wave][v + 8 * hi][nt * 16 + lan] = (__bf16)sc[nt][v];
        v16bf pa[2];
#pragma unroll
        for (int ks = 0; ks < 2; ++ks) {
            const int base = ks * 32 + 16 * hi;
            v8bf lo = *(const v8bf*)&p_lds[wave][lan][base];
            v8bf up = *(const v8bf*)&p_lds[wave][lan][base + 8];
#pragma unroll
            for (int e = 0; e < 8; ++e) { pa[ks][e] = lo[e]; pa[ks][e + 8] = up[e]; }
        }

        // ---- acc += P @ V over four 16-wide d-tiles, K=64 over 2 steps ----
#pragma unroll
        for (int t = 0; t < 4; ++t)
#pragma unroll
            for (int ks = 0; ks < 2; ++ks) {
                v16bf b = *(const v16bf*)&kvT_lds[cur][t * 16 + lan][ks * 32 + 16 * hi];
                acc[t] = __builtin_amdgcn_wmma_f32_16x16x32_bf16(
                             false, pa[ks], false, b, (short)0, acc[t], false, false);
            }

        // ---- store next tile into the other buffer; single barrier per iter ----
        if (have_next) {
            v16bf hv;
#pragma unroll
            for (int j = 0; j < 4; ++j)
#pragma unroll
                for (int e = 0; e < 4; ++e) hv[j * 4 + e] = (__bf16)nxt[j][e];
            *(v16bf*)&kv_lds[cur ^ 1][ld_r][ld_c] = hv;
#pragma unroll
            for (int e = 0; e < 16; ++e) kvT_lds[cur ^ 1][ld_c + e][ld_r] = hv[e];
        }
        __syncthreads();
    }

    // ---- epilogue: out = acc / l ----
#pragma unroll
    for (int v = 0; v < 8; ++v) {
        float inv = 1.0f / l[v];
        size_t ro = (size_t)(q0 + v + 8 * hi) * D_DIM;
#pragma unroll
        for (int t = 0; t < 4; ++t)
            og[ro + t * 16 + lan] = acc[t][v] * inv;
    }
}

extern "C" void kernel_launch(void* const* d_in, const int* in_sizes, int n_in,
                              void* d_out, int out_size, void* d_ws, size_t ws_size,
                              hipStream_t stream) {
    const float* q    = (const float*)d_in[0];
    const float* kv   = (const float*)d_in[1];
    const float* mask = (const float*)d_in[2];
    float* out        = (float*)d_out;

    dim3 grid(4 * 8 * (S_LEN / 128));  // 512 blocks: one per (b,h,128-row q tile)
    dim3 block(256);                   // 8 wave32s
    hipLaunchKernelGGL(fattn_fwd_kernel, grid, block, 0, stream, q, kv, mask, out);
}
